// UMA_71347996721585
// MI455X (gfx1250) — matrix operand order repaired and checked by reading, not compiled, not run on hardware
//
#include <hip/hip_runtime.h>
#include <math.h>

typedef __attribute__((ext_vector_type(16))) _Float16 v16h;
typedef __attribute__((ext_vector_type(8)))  float    v8f;

#define B_     16
#define L_     2048
#define D_     256
#define NPOS   126          // forced boundary positions: 24 + 16k, k=0..125
#define NCHUNK 127          // NPOS + 1 chunk bins
#define MS     1152         // MAX_SEGS

// ---------------------------------------------------------------------------
// Kernel 1: w = sigmoid(X @ w_lin + b) via v_wmma_f32_16x16x32_f16.
// One wave handles 16 frames; block = 256 threads = 8 waves = 128 frames.
// A (16x32, f16): lane m (0-15): halves[0..7]=K 0..7, halves[8..15]=K 16..23;
//                 lane m+16:     halves[0..7]=K 8..15, halves[8..15]=K 24..31.
// B (32x16, f16): w_lin chunk replicated in every column:
//                 lanes 0-15 hold K=0..7,16..23, lanes 16-31 hold K=8..15,24..31.
// C column 0: lane 0 -> rows 0..7 (vgpr g = row g), lane 16 -> rows 8..15.
// ---------------------------------------------------------------------------
__global__ __launch_bounds__(256)
void uma_proj_wmma(const float* __restrict__ xs, const float* __restrict__ w_lin,
                   const float* __restrict__ b_lin, float* __restrict__ w_out)
{
    __shared__ _Float16 wl16[D_];
    const int tid = threadIdx.x;
    if (tid < D_) wl16[tid] = (_Float16)w_lin[tid];
    __syncthreads();

    const int wave = tid >> 5;
    const int lane = tid & 31;
    const int hi   = lane >> 4;         // half-lane group (0 or 1)
    const int m    = lane & 15;         // A-matrix row owned by this lane
    const long frameBase = (long)blockIdx.x * 128 + (long)wave * 16;
    const float bias = b_lin[0];

    v8f c = {};
    for (int kc = 0; kc < D_; kc += 32) {
        const float* row = xs + (frameBase + m) * D_ + kc;
        v16h a;
        #pragma unroll
        for (int h = 0; h < 8; ++h) {
            a[h]     = (_Float16)row[8 * hi + h];        // K = 8*hi + h
            a[h + 8] = (_Float16)row[16 + 8 * hi + h];   // K = 16 + 8*hi + h
        }
        v16h bm;
        #pragma unroll
        for (int h = 0; h < 16; ++h)
            bm[h] = wl16[kc + (h < 8 ? 8 * hi + h : 16 + 8 * hi + (h - 8))];              // replicated columns
        c = __builtin_amdgcn_wmma_f32_16x16x32_f16(false, a, false, bm,
                                                   (short)0, c, false, false);
    }
    if (m == 0) {   // lane 0 -> rows 0..7, lane 16 -> rows 8..15
        #pragma unroll
        for (int g = 0; g < 8; ++g) {
            float s = c[g] + bias;
            w_out[frameBase + 8 * hi + g] = 1.0f / (1.0f + __expf(-s));
        }
    }
}

// ---------------------------------------------------------------------------
// Kernel 2: per-batch mask -> starts/ends lists, counts, chunk_counts.
// One block (256 threads) per batch item; each thread owns 8 frames.
// ---------------------------------------------------------------------------
__global__ __launch_bounds__(256)
void uma_segment(const float* __restrict__ w_buf,
                 int* __restrict__ starts, int* __restrict__ ends,
                 int* __restrict__ out_int)
{
    __shared__ int cntM[256], cntE[256], excM[257], excE[257];
    __shared__ int cc[NCHUNK];

    const int b   = blockIdx.x;
    const int tid = threadIdx.x;
    const float* wr = w_buf + (long)b * L_;

    if (tid < NCHUNK) cc[tid] = 0;

    unsigned mbits = 0, ebits = 0;
    const int t0 = tid * 8;
    #pragma unroll
    for (int j = 0; j < 8; ++j) {
        const int t = t0 + j;
        const float wd = wr[t];
        const float pv = (t > 0)      ? wr[t - 1] : 0.0f;   // F.pad zeros
        const float nx = (t < L_ - 1) ? wr[t + 1] : 0.0f;
        bool msk = (wd < pv) && (wd < nx);
        if (t == 0) msk = true;
        if (t >= 24 && t <= 24 + 16 * (NPOS - 1) && ((t - 24) & 15) == 0) msk = true;
        bool emk = msk;
        if (t == 0)      emk = false;
        if (t == L_ - 1) emk = true;
        if (msk) mbits |= 1u << j;
        if (emk) ebits |= 1u << j;
    }
    cntM[tid] = __popc(mbits);
    cntE[tid] = __popc(ebits);
    __syncthreads();
    if (tid == 0) {              // 256-element serial scan: negligible cost
        int sm = 0, se = 0;
        for (int i = 0; i < 256; ++i) {
            excM[i] = sm; sm += cntM[i];
            excE[i] = se; se += cntE[i];
        }
        excM[256] = sm; excE[256] = se;
    }
    __syncthreads();

    int r = excM[tid], re = excE[tid];
    #pragma unroll
    for (int j = 0; j < 8; ++j) {
        const int t = t0 + j;
        if (mbits & (1u << j)) {
            if (r < MS) starts[(long)b * MS + r] = t;
            ++r;
            const int seg = (t < 24) ? 0 : min(NPOS, (t - 24) / 16 + 1);
            atomicAdd(&cc[seg], 1);
        }
        if (ebits & (1u << j)) {
            if (re < MS) ends[(long)b * MS + re] = min(t + 2, L_);
            ++re;
        }
    }
    __syncthreads();
    const int totM = excM[256], totE = excE[256];
    for (int i = totM + tid; i < MS; i += 256) starts[(long)b * MS + i] = L_;
    for (int i = totE + tid; i < MS; i += 256) ends[(long)b * MS + i]   = L_;
    if (tid == 0)      out_int[b] = totM;                       // counts
    if (tid < NCHUNK)  out_int[B_ + b * NCHUNK + tid] = cc[tid];// chunk_counts
}

// ---------------------------------------------------------------------------
// Kernel 3: per-segment weighted pooling + LayerNorm. One wave per segment
// (all MS rows; empty/padded segments produce beta, matching the reference).
// Lane owns 8 contiguous dims -> wave reads a full 1024B row coalesced.
// ---------------------------------------------------------------------------
__global__ __launch_bounds__(256)
void uma_pool_ln(const float* __restrict__ xs, const float* __restrict__ w_buf,
                 const int* __restrict__ starts, const int* __restrict__ ends,
                 const float* __restrict__ gamma, const float* __restrict__ beta,
                 float* __restrict__ out)
{
    const int tid  = threadIdx.x;
    const int lane = tid & 31;
    const long seg = (long)blockIdx.x * 8 + (tid >> 5);
    const int b = (int)(seg / MS);
    int s = starts[seg];
    int e = ends[seg];
    s = max(0, min(s, L_));
    e = max(0, min(e, L_));

    float acc[8] = {0.f,0.f,0.f,0.f,0.f,0.f,0.f,0.f};
    float den = 0.0f;
    for (int t = s; t < e; ++t) {
        const float wt = w_buf[(long)b * L_ + t];
        const float* xr = xs + ((long)b * L_ + t) * D_ + lane * 8;
        #pragma unroll
        for (int j = 0; j < 8; ++j) acc[j] = fmaf(wt, xr[j], acc[j]);
        den += wt;
    }
    const float inv = 1.0f / fmaxf(den, 1e-6f);
    float o[8]; float lsum = 0.0f;
    #pragma unroll
    for (int j = 0; j < 8; ++j) { o[j] = acc[j] * inv; lsum += o[j]; }
    #pragma unroll
    for (int off = 16; off > 0; off >>= 1) lsum += __shfl_xor(lsum, off, 32);
    const float mu = lsum * (1.0f / D_);
    float lvar = 0.0f;
    #pragma unroll
    for (int j = 0; j < 8; ++j) { const float d = o[j] - mu; lvar += d * d; }
    #pragma unroll
    for (int off = 16; off > 0; off >>= 1) lvar += __shfl_xor(lvar, off, 32);
    const float rs = rsqrtf(lvar * (1.0f / D_) + 1e-5f);
    float* op = out + seg * D_ + lane * 8;
    #pragma unroll
    for (int j = 0; j < 8; ++j)
        op[j] = (o[j] - mu) * rs * gamma[lane * 8 + j] + beta[lane * 8 + j];
}

// ---------------------------------------------------------------------------
extern "C" void kernel_launch(void* const* d_in, const int* in_sizes, int n_in,
                              void* d_out, int out_size, void* d_ws, size_t ws_size,
                              hipStream_t stream)
{
    const float* xs    = (const float*)d_in[0];
    /* d_in[1] = olens (int32) unused: always L */
    const float* w_lin = (const float*)d_in[2];
    const float* b_lin = (const float*)d_in[3];
    const float* gamma = (const float*)d_in[4];
    const float* beta  = (const float*)d_in[5];

    float* w_buf = (float*)d_ws;                          // B*L floats
    int*   starts = (int*)(w_buf + (size_t)B_ * L_);      // B*MS ints
    int*   ends   = starts + (size_t)B_ * MS;             // B*MS ints

    float* out_f = (float*)d_out;                         // [B, MS, D] floats
    int*   out_i = (int*)(out_f + (size_t)B_ * MS * D_);  // counts + chunk_counts

    uma_proj_wmma<<<(B_ * L_) / 128, 256, 0, stream>>>(xs, w_lin, b_lin, w_buf);
    uma_segment  <<<B_,              256, 0, stream>>>(w_buf, starts, ends, out_i);
    uma_pool_ln  <<<(B_ * MS) / 8,   256, 0, stream>>>(xs, w_buf, starts, ends,
                                                       gamma, beta, out_f);
}
